// SGFormerSeed_360777253125
// MI455X (gfx1250) — compile-verified, matrix operand order checked by reading, not run on hardware
//
#include <hip/hip_runtime.h>
#include <hip/hip_bf16.h>
#include <math.h>

// ---------------------------------------------------------------------------
// SGFormerSeed forward for MI455X (gfx1250, wave32, WMMA).
// Math simplifications vs reference:
//  * softmax over a single key == 1.0  -> Wq/Wk GEMMs are dead, attention
//    output is a constant row v@Wo+bo (two 128x128 matvecs on the seed row).
//  * seed_emb concat block is a constant row -> folded into an effective bias
//    for the final GEMM; final GEMM = 3 K=128 slices (h0 slice weights are
//    fc_W[128:256]+fc_W[512:640] pre-summed during LDS staging).
// All GEMMs: bf16 inputs (RNE-converted in LDS), f32 accumulation via
// v_wmma_f32_16x16x32_bf16. Everything else is memory-path code (atomics for
// segment sums / BN stats, wave shuffles for LN).
// ---------------------------------------------------------------------------

#define EPS_F 1e-5f

typedef __attribute__((ext_vector_type(16))) __bf16 v16bf;
typedef __attribute__((ext_vector_type(8)))  float  v8f;
typedef __attribute__((ext_vector_type(4)))  unsigned int u32x4;
typedef __attribute__((ext_vector_type(4)))  float  f32x4;

__device__ __forceinline__ unsigned short f2bf(float f) {
  unsigned int u = __builtin_bit_cast(unsigned int, f);
  unsigned int r = u + 0x7FFFu + ((u >> 16) & 1u);   // round-to-nearest-even
  return (unsigned short)(r >> 16);
}

// Load one 16x32 bf16 fragment (A or transposed-B) from LDS, row-major
// [16 rows x 128 k], following the CDNA5 16-bit A-matrix VGPR layout:
// lane L: row = L&15; lanes 0-15 hold k = kk*32 + [0..7, 16..23],
// lanes 16-31 hold k = kk*32 + [8..15, 24..31]  -> two contiguous 16B chunks.
__device__ __forceinline__ v16bf load_frag(const unsigned short* lds,
                                           int rowBase, int kk) {
  const int lane = threadIdx.x & 31;
  const unsigned short* p =
      lds + (rowBase + (lane & 15)) * 128 + kk * 32 + (lane >> 4) * 8;
  union { u32x4 q[2]; v16bf v; } f;
  f.q[0] = *(const u32x4*)p;        // elements 0..7  (k .. k+7)
  f.q[1] = *(const u32x4*)(p + 16); // elements 8..15 (k+16 .. k+23)
  return f.v;
}

// ---------------------------------------------------------------------------
// C[M,128] = A[M,128] @ W[128,128] (+ bias).  Block: 256 thr (8 waves),
// tile 64 rows x 128 cols; wave w owns column tile w, loops 4 row subtiles.
// ---------------------------------------------------------------------------
__global__ __launch_bounds__(256)
void gemm128_kernel(const float* __restrict__ A, const float* __restrict__ W,
                    const float* __restrict__ bias, float* __restrict__ C,
                    int M) {
  __shared__ __align__(16) unsigned short ldsA[64 * 128];   // 16 KB
  __shared__ __align__(16) unsigned short ldsW[128 * 128];  // 32 KB (transposed)
  const int tid  = threadIdx.x;
  const int base = blockIdx.x * 64;

  for (int i = 0; i < 32; ++i) {                 // A tile (coalesced)
    int idx = tid + i * 256;
    int r = idx >> 7, k = idx & 127;
    int row = base + r;
    ldsA[idx] = f2bf(row < M ? A[(size_t)row * 128 + k] : 0.0f);
  }
  for (int i = 0; i < 64; ++i) {                 // W transposed: ldsW[n*128+k]
    int idx = tid + i * 256;
    int k = idx >> 7, n = idx & 127;             // coalesced global read
    ldsW[n * 128 + k] = f2bf(W[(size_t)k * 128 + n]);
  }
  __syncthreads();

  const int wave = tid >> 5;                     // column tile 0..7
  v8f acc[4];
  for (int i = 0; i < 4; ++i) acc[i] = (v8f)0.0f;

  for (int kk = 0; kk < 4; ++kk) {
    v16bf bf = load_frag(ldsW, wave * 16, kk);
    for (int mi = 0; mi < 4; ++mi) {
      v16bf af = load_frag(ldsA, mi * 16, kk);
      acc[mi] = __builtin_amdgcn_wmma_f32_16x16x32_bf16(
          false, af, false, bf, (short)0, acc[mi], false, false);
    }
  }

  const int lane = tid & 31, half = lane >> 4;
  const int col  = wave * 16 + (lane & 15);
  const float bc = bias ? bias[col] : 0.0f;
  for (int mi = 0; mi < 4; ++mi)
    for (int r = 0; r < 8; ++r) {
      int row = base + mi * 16 + half * 8 + r;   // C layout: VGPR r -> M=r/r+8
      if (row < M) C[(size_t)row * 128 + col] = acc[mi][r] + bc;
    }
}

// ---------------------------------------------------------------------------
// Final GEMM: out[M,64] = tanh(A0@W[0:128] + A1@(W[128:256]+W[512:640])
//                              + A2@W[256:384] + ebias)
// Block: 128 thr (4 waves), tile 32 rows x 64 cols, 3 K=128 passes.
// ---------------------------------------------------------------------------
__global__ __launch_bounds__(128)
void final_gemm_kernel(const float* __restrict__ A0, const float* __restrict__ A1,
                       const float* __restrict__ A2, const float* __restrict__ fcW,
                       const float* __restrict__ ebias, float* __restrict__ out,
                       int M) {
  __shared__ __align__(16) unsigned short ldsA[32 * 128];  // 8 KB
  __shared__ __align__(16) unsigned short ldsW[64 * 128];  // 16 KB
  const int tid  = threadIdx.x;
  const int base = blockIdx.x * 32;
  const int wave = tid >> 5;                     // column tile 0..3

  v8f acc[2];
  acc[0] = (v8f)0.0f; acc[1] = (v8f)0.0f;

  for (int s = 0; s < 3; ++s) {
    const float* src = (s == 0) ? A0 : (s == 1) ? A1 : A2;
    const int rowbase = (s == 0) ? 0 : (s == 1) ? 128 : 256;
    for (int i = 0; i < 32; ++i) {               // A tile
      int idx = tid + i * 128;
      int r = idx >> 7, k = idx & 127;
      int row = base + r;
      ldsA[idx] = f2bf(row < M ? src[(size_t)row * 128 + k] : 0.0f);
    }
    for (int i = 0; i < 64; ++i) {               // W slice transposed
      int idx = tid + i * 128;
      int n = idx >> 7, k = idx & 127;
      float w = fcW[(size_t)(rowbase + k) * 64 + n];
      if (s == 1) w += fcW[(size_t)(512 + k) * 64 + n];  // seed fold partner
      ldsW[n * 128 + k] = f2bf(w);
    }
    __syncthreads();
    for (int kk = 0; kk < 4; ++kk) {
      v16bf bf = load_frag(ldsW, wave * 16, kk);
      for (int mi = 0; mi < 2; ++mi) {
        v16bf af = load_frag(ldsA, mi * 16, kk);
        acc[mi] = __builtin_amdgcn_wmma_f32_16x16x32_bf16(
            false, af, false, bf, (short)0, acc[mi], false, false);
      }
    }
    __syncthreads();
  }

  const int lane = tid & 31, half = lane >> 4;
  const int col  = wave * 16 + (lane & 15);
  const float eb = ebias[col];
  for (int mi = 0; mi < 2; ++mi)
    for (int r = 0; r < 8; ++r) {
      int row = base + mi * 16 + half * 8 + r;
      if (row < M) out[(size_t)row * 64 + col] = tanhf(acc[mi][r] + eb);
    }
}

// ------------------------- small / memory-path kernels ----------------------

__global__ void zero_kernel(float* __restrict__ p, int n) {
  for (int i = blockIdx.x * blockDim.x + threadIdx.x; i < n;
       i += gridDim.x * blockDim.x)
    p[i] = 0.0f;
}

// column sums / sums-of-squares for BatchNorm (atomics into stats[0:256])
__global__ __launch_bounds__(128)
void colstats_kernel(const float* __restrict__ y, float* __restrict__ stats,
                     int M) {
  const int c = threadIdx.x;  // column 0..127
  float s = 0.0f, s2 = 0.0f;
  for (int r = blockIdx.x; r < M; r += gridDim.x) {
    float v = y[(size_t)r * 128 + c];
    s += v; s2 += v * v;
  }
  atomicAdd(&stats[c], s);
  atomicAdd(&stats[128 + c], s2);
}

// out = relu((y-mean)*g/sqrt(var+eps)+b) (+ res)   [elementwise, float4]
__global__ __launch_bounds__(256)
void bn_relu_kernel(const float* __restrict__ y, const float* __restrict__ stats,
                    const float* __restrict__ g, const float* __restrict__ b,
                    const float* __restrict__ res, float* __restrict__ out,
                    int M) {
  const int total = M * 32;                      // float4 count
  const float invM = 1.0f / (float)M;
  for (int i = blockIdx.x * blockDim.x + threadIdx.x; i < total;
       i += gridDim.x * blockDim.x) {
    int c0 = (i & 31) * 4;
    f32x4 v = *(const f32x4*)(y + (size_t)i * 4);
    f32x4 o;
    for (int j = 0; j < 4; ++j) {
      int c = c0 + j;
      float mean = stats[c] * invM;
      float var  = stats[128 + c] * invM - mean * mean;
      float sc   = g[c] * rsqrtf(var + EPS_F);
      o[j] = fmaxf((v[j] - mean) * sc + b[c], 0.0f);
    }
    if (res) o += *(const f32x4*)(res + (size_t)i * 4);
    *(f32x4*)(out + (size_t)i * 4) = o;
  }
}

// row LayerNorm + relu; one wave per row (4 floats/lane, shfl_xor reduce).
// If mix != nullptr: input = 0.5*mix[col] + 0.5*y (the ALPHA residual mix).
__global__ __launch_bounds__(256)
void ln_relu_kernel(const float* __restrict__ y, const float* __restrict__ mix,
                    const float* __restrict__ g, const float* __restrict__ b,
                    float* __restrict__ out, int M) {
  const int gtid = blockIdx.x * blockDim.x + threadIdx.x;
  const int lane = gtid & 31;
  const int nw   = (gridDim.x * blockDim.x) >> 5;
  for (int row = gtid >> 5; row < M; row += nw) {
    f32x4 v = *(const f32x4*)(y + (size_t)row * 128 + lane * 4);
    if (mix) {
      f32x4 mc = *(const f32x4*)(mix + lane * 4);
      v = 0.5f * v + 0.5f * mc;
    }
    float s = v[0] + v[1] + v[2] + v[3];
    for (int o = 16; o; o >>= 1) s += __shfl_xor(s, o, 32);
    float mean = s * (1.0f / 128.0f);
    f32x4 d = v - mean;
    float s2 = d[0] * d[0] + d[1] * d[1] + d[2] * d[2] + d[3] * d[3];
    for (int o = 16; o; o >>= 1) s2 += __shfl_xor(s2, o, 32);
    float rstd = rsqrtf(s2 * (1.0f / 128.0f) + EPS_F);
    f32x4 gg = *(const f32x4*)(g + lane * 4);
    f32x4 bb = *(const f32x4*)(b + lane * 4);
    f32x4 o4 = d * rstd * gg + bb;
    for (int j = 0; j < 4; ++j) o4[j] = fmaxf(o4[j], 0.0f);
    *(f32x4*)(out + (size_t)row * 128 + lane * 4) = o4;
  }
}

// per-type in-degree (atomic) ; edge_index row-major [2,E]
__global__ void deg_kernel(const int* __restrict__ ei, const int* __restrict__ et,
                           float* __restrict__ deg0, float* __restrict__ deg1,
                           int E) {
  for (int e = blockIdx.x * blockDim.x + threadIdx.x; e < E;
       e += gridDim.x * blockDim.x) {
    int c = ei[E + e];
    atomicAdd(((et[e] == 0) ? deg0 : deg1) + c, 1.0f);
  }
}

__global__ void dnorm_kernel(float* __restrict__ p, int n) {  // in-place
  for (int i = blockIdx.x * blockDim.x + threadIdx.x; i < n;
       i += gridDim.x * blockDim.x) {
    float d = p[i];
    p[i] = (d > 0.0f) ? rsqrtf(d) : 0.0f;
  }
}

// SpMM scatter: agg[col] += w*dn[row]*dn[col]*h[row]; one wave per edge,
// 4 features per lane, value computed inline.
__global__ __launch_bounds__(256)
void scatter_kernel(const float* __restrict__ h, const float* __restrict__ dn,
                    const int* __restrict__ ei, const int* __restrict__ et,
                    const float* __restrict__ ew, int type,
                    float* __restrict__ agg, int E) {
  const int lane = threadIdx.x & 31;
  const int nw   = (gridDim.x * blockDim.x) >> 5;
  for (int e = (blockIdx.x * blockDim.x + threadIdx.x) >> 5; e < E; e += nw) {
    if (et[e] != type) continue;
    int r = ei[e], c = ei[E + e];
    float v = ew[e] * dn[r] * dn[c];
    if (v == 0.0f) continue;
    f32x4 hv = *(const f32x4*)(h + (size_t)r * 128 + lane * 4);
    float* dst = agg + (size_t)c * 128 + lane * 4;
    atomicAdd(dst + 0, v * hv[0]);
    atomicAdd(dst + 1, v * hv[1]);
    atomicAdd(dst + 2, v * hv[2]);
    atomicAdd(dst + 3, v * hv[3]);
  }
}

// attention constant row: outc = (hT[seed]@Wv+bv)@Wo+bo   (<<<1,128>>>)
__global__ __launch_bounds__(128)
void seed_const_kernel(const float* __restrict__ hT, const int* __restrict__ seedp,
                       const float* __restrict__ Wv, const float* __restrict__ bv,
                       const float* __restrict__ Wo, const float* __restrict__ bo,
                       float* __restrict__ outc) {
  __shared__ float s[128], v[128];
  const int t = threadIdx.x;
  const int seed = *seedp;
  s[t] = hT[(size_t)seed * 128 + t];
  __syncthreads();
  float a = bv[t];
  for (int k = 0; k < 128; ++k) a += s[k] * Wv[(size_t)k * 128 + t];
  v[t] = a;
  __syncthreads();
  float a2 = bo[t];
  for (int k = 0; k < 128; ++k) a2 += v[k] * Wo[(size_t)k * 128 + t];
  outc[t] = a2;
}

// effective bias: ebias = fc_b + h0[seed] @ fc_W[384:512]   (<<<1,64>>>)
__global__ __launch_bounds__(64)
void final_bias_kernel(const float* __restrict__ h0, const int* __restrict__ seedp,
                       const float* __restrict__ fcW, const float* __restrict__ fcb,
                       float* __restrict__ ebias) {
  const int n = threadIdx.x;
  const int seed = *seedp;
  float a = fcb[n];
  for (int k = 0; k < 128; ++k)
    a += h0[(size_t)seed * 128 + k] * fcW[(size_t)(384 + k) * 64 + n];
  ebias[n] = a;
}

// ---------------------------------------------------------------------------
extern "C" void kernel_launch(void* const* d_in, const int* in_sizes, int n_in,
                              void* d_out, int out_size, void* d_ws, size_t ws_size,
                              hipStream_t stream) {
  const int N = in_sizes[0] / 128;   // nodes
  const int E = in_sizes[2];         // edges

  // inputs (JAX pytree / sorted-dict-key flattening of params assumed)
  const float* x     = (const float*)d_in[0];
  const int*   ei    = (const int*)d_in[1];
  const int*   et    = (const int*)d_in[2];
  const float* ew    = (const float*)d_in[3];
  const int*   seedp = (const int*)d_in[4];
  const float* fcW   = (const float*)d_in[5];
  const float* fcb   = (const float*)d_in[6];
  // g0: W0,b0,bn0_b,bn0_g, layer{W,b,bn_b,bn_g}
  const float* g0W0 = (const float*)d_in[7],  *g0b0 = (const float*)d_in[8];
  const float* g0nb = (const float*)d_in[9],  *g0ng = (const float*)d_in[10];
  const float* g0W  = (const float*)d_in[11], *g0b  = (const float*)d_in[12];
  const float* g0lb = (const float*)d_in[13], *g0lg = (const float*)d_in[14];
  // g1
  const float* g1W0 = (const float*)d_in[15], *g1b0 = (const float*)d_in[16];
  const float* g1nb = (const float*)d_in[17], *g1ng = (const float*)d_in[18];
  const float* g1W  = (const float*)d_in[19], *g1b  = (const float*)d_in[20];
  const float* g1lb = (const float*)d_in[21], *g1lg = (const float*)d_in[22];
  // trans: in_W,in_b, layer{Wk,Wo,Wq,Wv,bk,bo,bq,bv,ln_b,ln_g}, ln0_b,ln0_g
  const float* inW  = (const float*)d_in[23], *inb  = (const float*)d_in[24];
  const float* Wo   = (const float*)d_in[26], *Wv   = (const float*)d_in[28];
  const float* bo   = (const float*)d_in[30], *bv   = (const float*)d_in[32];
  const float* lnb  = (const float*)d_in[33], *lng  = (const float*)d_in[34];
  const float* ln0b = (const float*)d_in[35], *ln0g = (const float*)d_in[36];
  (void)n_in; (void)out_size; (void)ws_size;

  // workspace layout (fp32): 4 N*128 buffers + 2N degree + stats/const
  float* ws    = (float*)d_ws;
  float* B0    = ws;                         // x_trans / hT
  float* B1    = B0 + (size_t)N * 128;       // h0
  float* B2    = B1 + (size_t)N * 128;       // h1
  float* B3    = B2 + (size_t)N * 128;       // gemm scratch / agg
  float* dn0   = B3 + (size_t)N * 128;       // deg->dnorm type0 [N]
  float* dn1   = dn0 + N;                    // deg->dnorm type1 [N]
  float* stats = dn1 + N;                    // [256] sum|sumsq
  float* outc  = stats + 256;                // [128] attn const row
  float* ebias = outc + 128;                 // [64]

  const int gemmGrid = (N + 63) / 64;
  float* out = (float*)d_out;

  // ---- transformer branch -------------------------------------------------
  gemm128_kernel<<<gemmGrid, 256, 0, stream>>>(x, inW, inb, B3, N);
  ln_relu_kernel<<<2048, 256, 0, stream>>>(B3, nullptr, ln0g, ln0b, B0, N);
  seed_const_kernel<<<1, 128, 0, stream>>>(B0, seedp, Wv, bv, Wo, bo, outc);
  ln_relu_kernel<<<2048, 256, 0, stream>>>(B0, outc, lng, lnb, B0, N); // in-place

  // ---- degrees / symmetric norm ------------------------------------------
  zero_kernel<<<1024, 256, 0, stream>>>(dn0, 2 * N);
  deg_kernel<<<4096, 256, 0, stream>>>(ei, et, dn0, dn1, E);
  dnorm_kernel<<<1024, 256, 0, stream>>>(dn0, 2 * N);

  // ---- graph conv branches ------------------------------------------------
  for (int t = 0; t < 2; ++t) {
    const float *W0 = t ? g1W0 : g0W0, *b0 = t ? g1b0 : g0b0;
    const float *ng = t ? g1ng : g0ng, *nb = t ? g1nb : g0nb;
    const float *W  = t ? g1W  : g0W,  *b  = t ? g1b  : g0b;
    const float *lg = t ? g1lg : g0lg, *lb = t ? g1lb : g0lb;
    const float *dn = t ? dn1 : dn0;
    float* H = t ? B2 : B1;

    gemm128_kernel<<<gemmGrid, 256, 0, stream>>>(x, W0, b0, B3, N);
    zero_kernel<<<1, 256, 0, stream>>>(stats, 256);
    colstats_kernel<<<512, 128, 0, stream>>>(B3, stats, N);
    bn_relu_kernel<<<2048, 256, 0, stream>>>(B3, stats, ng, nb, nullptr, H, N);

    zero_kernel<<<4096, 256, 0, stream>>>(B3, N * 128);
    scatter_kernel<<<16384, 256, 0, stream>>>(H, dn, ei, et, ew, t, B3, E);

    gemm128_kernel<<<gemmGrid, 256, 0, stream>>>(B3, W, b, B3, N); // in-place ok
    zero_kernel<<<1, 256, 0, stream>>>(stats, 256);
    colstats_kernel<<<512, 128, 0, stream>>>(B3, stats, N);
    bn_relu_kernel<<<2048, 256, 0, stream>>>(B3, stats, lg, lb, H, H, N); // +res
  }

  // ---- fused output GEMM (seed block folded into bias) --------------------
  final_bias_kernel<<<1, 64, 0, stream>>>(B1, seedp, fcW, fcb, ebias);
  final_gemm_kernel<<<(N + 31) / 32, 128, 0, stream>>>(B0, B1, B2, fcW, ebias,
                                                       out, N);
}
